// MCLETLayer_28037546509014
// MI455X (gfx1250) — matrix operand order repaired and checked by reading, not compiled, not run on hardware
//
#include <hip/hip_runtime.h>

#define N_NODES 50000
#define N_EDGES 800000
#define DIM2    256
#define N_TYPES 256
#define N_HEADS 4
#define NSTRIP  (N_EDGES / 16)
#define LDS_STRIDE 264   // halves per W row: 256 + 8 pad -> conflict-free ds_load_b128

typedef __attribute__((ext_vector_type(16))) _Float16 v16h;
typedef __attribute__((ext_vector_type(8)))  _Float16 v8h;
typedef __attribute__((ext_vector_type(8)))  float    v8f;

// ---- monotone float<->uint key for atomic segment-max ----
__device__ __forceinline__ unsigned fkey(float f) {
    unsigned b = __float_as_uint(f);
    return (b & 0x80000000u) ? ~b : (b | 0x80000000u);
}
__device__ __forceinline__ float fdec(unsigned k) {
    unsigned b = (k & 0x80000000u) ? (k ^ 0x80000000u) : ~k;
    return __uint_as_float(b);
}

__global__ void zero_kernel(float* p, size_t n) {
    size_t i  = (size_t)blockIdx.x * blockDim.x + threadIdx.x;
    size_t st = (size_t)gridDim.x * blockDim.x;
    for (; i < n; i += st) p[i] = 0.0f;
}

// ---- cooperative W_fc (f32) -> LDS (f16, padded stride) ----
__device__ __forceinline__ void fill_w_lds(_Float16* wl, const float* __restrict__ W) {
    for (int i = threadIdx.x; i < N_TYPES * DIM2; i += blockDim.x) {
        int t = i >> 8, k = i & 255;
        wl[t * LDS_STRIDE + k] = (_Float16)W[i];
    }
    __syncthreads();
}

__device__ __forceinline__ void ld8(float* dst, const float* __restrict__ p) {
    float4 a = *(const float4*)p;
    float4 b = *(const float4*)(p + 4);
    dst[0] = a.x; dst[1] = a.y; dst[2] = a.z; dst[3] = a.w;
    dst[4] = b.x; dst[5] = b.y; dst[6] = b.z; dst[7] = b.w;
}

// A fragment (16x32 f16 per K-step): lane = row (lane&15); lanes<16 hold K {0-7,16-23},
// lanes>=16 hold K {8-15,24-31} (CDNA5 ISA 7.12.2).  msg = relu(src+edge) computed on the fly.
__device__ __forceinline__ void load_a(v16h a[8], const float* __restrict__ srcrow,
                                       const float* __restrict__ edgerow, int hi) {
#pragma unroll
    for (int ks = 0; ks < 8; ++ks) {
        int kb0 = ks * 32 + hi * 8;
        int kb1 = kb0 + 16;
        float s0[8], s1[8], e0[8], e1[8];
        ld8(s0, srcrow + kb0);  ld8(e0, edgerow + kb0);
        ld8(s1, srcrow + kb1);  ld8(e1, edgerow + kb1);
        v16h av;
#pragma unroll
        for (int i = 0; i < 8; ++i) {
            av[i]     = (_Float16)fmaxf(s0[i] + e0[i], 0.0f);
            av[i + 8] = (_Float16)fmaxf(s1[i] + e1[i], 0.0f);
        }
        a[ks] = av;
    }
}

// B fragment (32x16 f16): lane = col (lane&15); lanes 0-15 hold K {0-7,16-23}, lanes 16-31 K {8-15,24-31}.
__device__ __forceinline__ v16h load_b(const _Float16* wl, int t, int kb) {
    const v8h* pb = (const v8h*)(wl + t * LDS_STRIDE + kb);
    v8h lo = pb[0], hi = pb[2];
    v16h b;
#pragma unroll
    for (int i = 0; i < 8; ++i) { b[i] = lo[i]; b[i + 8] = hi[i]; }
    return b;
}

// ===== Pass 1: GEMM -> per-edge scores [E,4] + atomic segment-max keys =====
__global__ __launch_bounds__(128) void edge_gemm_scores(
    const float* __restrict__ src, const float* __restrict__ edge,
    const float* __restrict__ Wfc, const float* __restrict__ bfc,
    const float* __restrict__ Q,
    const int* __restrict__ src_idx, const int* __restrict__ dst_idx,
    float* __restrict__ scores, unsigned* __restrict__ smax_keys) {
    extern __shared__ _Float16 wl[];
    fill_w_lds(wl, Wfc);
    int lane = threadIdx.x & 31;
    int wid  = threadIdx.x >> 5;
    int hi   = lane >> 4;
    int nloc = lane & 15;

    for (int s = blockIdx.x * 4 + wid; s < NSTRIP; s += gridDim.x * 4) {
        int e0 = s * 16;
        int arow = e0 + nloc;                       // A row carried by this lane
        int srow = src_idx[arow];
        v16h a[8];
        load_a(a, src + (size_t)srow * DIM2, edge + (size_t)arow * DIM2, hi);

        float sc[8][4];
#pragma unroll
        for (int j = 0; j < 8; ++j)
#pragma unroll
            for (int h = 0; h < 4; ++h) sc[j][h] = 0.0f;

        for (int nt = 0; nt < 16; ++nt) {
            v8f acc = {};
#pragma unroll
            for (int ks = 0; ks < 8; ++ks) {
                v16h b = load_b(wl, nt * 16 + nloc, ks * 32 + hi * 8);
                acc = __builtin_amdgcn_wmma_f32_16x16x32_f16(
                    false, a[ks], false, b, (short)0, acc, false, false);
            }
            int col = nt * 16 + nloc;
            float bc = bfc[col];
            float q0 = Q[col], q1 = Q[256 + col], q2 = Q[512 + col], q3 = Q[768 + col];
#pragma unroll
            for (int j = 0; j < 8; ++j) {
                float pv = acc[j] + bc;             // p[row=j+hi*8][col]
                sc[j][0] += pv * q0; sc[j][1] += pv * q1;
                sc[j][2] += pv * q2; sc[j][3] += pv * q3;
            }
        }
        // reduce over 16-lane half (sum over columns)
#pragma unroll
        for (int j = 0; j < 8; ++j)
#pragma unroll
            for (int h = 0; h < 4; ++h) {
                float v = sc[j][h];
                v += __shfl_xor(v, 1, 32);
                v += __shfl_xor(v, 2, 32);
                v += __shfl_xor(v, 4, 32);
                v += __shfl_xor(v, 8, 32);
                sc[j][h] = v;
            }
        if (nloc < 8) {                              // lane nloc handles row = hi*8+nloc
            int row = hi * 8 + nloc;
            int e = e0 + row;
            int d = dst_idx[e];
#pragma unroll
            for (int h = 0; h < 4; ++h) {
                float v = sc[nloc][h];
                scores[(size_t)e * 4 + h] = v;
                atomicMax(smax_keys + (size_t)d * 4 + h, fkey(v));
            }
        }
    }
}

// ===== Pass 2: GEMM again -> softmax weights + atomic scatter of pooled/sum_p/denom/deg =====
__global__ __launch_bounds__(128) void edge_gemm_scatter(
    const float* __restrict__ src, const float* __restrict__ edge,
    const float* __restrict__ Wfc, const float* __restrict__ bfc,
    const int* __restrict__ src_idx, const int* __restrict__ dst_idx,
    const float* __restrict__ scores, const unsigned* __restrict__ smax_keys,
    float* __restrict__ denom, float* __restrict__ deg,
    float* __restrict__ sump, float* __restrict__ pooled) {
    extern __shared__ _Float16 wl[];
    fill_w_lds(wl, Wfc);
    int lane = threadIdx.x & 31;
    int wid  = threadIdx.x >> 5;
    int hi   = lane >> 4;
    int nloc = lane & 15;

    for (int s = blockIdx.x * 4 + wid; s < NSTRIP; s += gridDim.x * 4) {
        int e0 = s * 16;
        int arow = e0 + nloc;
        int srow = src_idx[arow];
        v16h a[8];
        load_a(a, src + (size_t)srow * DIM2, edge + (size_t)arow * DIM2, hi);

        // per-lane: softmax numerators for this half's 8 rows
        float w8[8][4];
        int   drow[8];
#pragma unroll
        for (int j = 0; j < 8; ++j) {
            int row = hi * 8 + j;
            int e = e0 + row;
            int d = dst_idx[e];
            drow[j] = d;
#pragma unroll
            for (int h = 0; h < 4; ++h) {
                float smax = fdec(smax_keys[(size_t)d * 4 + h]);
                w8[j][h] = __expf(scores[(size_t)e * 4 + h] - smax);
            }
        }
        if (nloc < 8) {                              // one lane per row: denom & degree
            int j = nloc;
            atomicAdd(deg + drow[j], 1.0f);
#pragma unroll
            for (int h = 0; h < 4; ++h)
                atomicAdd(denom + (size_t)drow[j] * 4 + h, w8[j][h]);
        }

        for (int nt = 0; nt < 16; ++nt) {
            v8f acc = {};
#pragma unroll
            for (int ks = 0; ks < 8; ++ks) {
                v16h b = load_b(wl, nt * 16 + nloc, ks * 32 + hi * 8);
                acc = __builtin_amdgcn_wmma_f32_16x16x32_f16(
                    false, a[ks], false, b, (short)0, acc, false, false);
            }
            int col = nt * 16 + nloc;
            float bc = bfc[col];
#pragma unroll
            for (int j = 0; j < 8; ++j) {
                float pv = acc[j] + bc;
                int d = drow[j];
                atomicAdd(sump + (size_t)d * 256 + col, pv);
#pragma unroll
                for (int h = 0; h < 4; ++h)
                    atomicAdd(pooled + ((size_t)d * 4 + h) * 256 + col, w8[j][h] * pv);
            }
        }
    }
}

// ===== Pass 3: per-node gate + normalized pooled -> sigmoid output =====
__global__ __launch_bounds__(256) void node_kernel(
    const float* __restrict__ Wg, const float* __restrict__ bg,
    const float* __restrict__ denom, const float* __restrict__ deg,
    const float* __restrict__ sump, const float* __restrict__ pooled,
    float* __restrict__ out) {
    int wid = threadIdx.x >> 5, lane = threadIdx.x & 31;
    int n = blockIdx.x * 8 + wid;
    if (n >= N_NODES) return;

    const float* sp = sump + (size_t)n * 256;
    float inv = 1.0f / fmaxf(deg[n], 1.0f);
    float part[4] = {0.f, 0.f, 0.f, 0.f};
#pragma unroll
    for (int i = 0; i < 8; ++i) {
        int t = lane * 8 + i;
        float mp = sp[t] * inv;
#pragma unroll
        for (int h = 0; h < 4; ++h) part[h] += mp * Wg[h * 256 + t];
    }
#pragma unroll
    for (int h = 0; h < 4; ++h) {
        float v = part[h];
#pragma unroll
        for (int off = 1; off < 32; off <<= 1) v += __shfl_xor(v, off, 32);
        part[h] = v + bg[h];
    }
    float m = fmaxf(fmaxf(part[0], part[1]), fmaxf(part[2], part[3]));
    float g[4], gs = 0.0f;
#pragma unroll
    for (int h = 0; h < 4; ++h) { g[h] = __expf(part[h] - m); gs += g[h]; }
    float gd[4];
#pragma unroll
    for (int h = 0; h < 4; ++h)
        gd[h] = (g[h] / gs) / fmaxf(denom[(size_t)n * 4 + h], 1e-9f);
#pragma unroll
    for (int i = 0; i < 8; ++i) {
        int t = lane * 8 + i;
        float sacc = 0.0f;
#pragma unroll
        for (int h = 0; h < 4; ++h)
            sacc += gd[h] * pooled[((size_t)n * 4 + h) * 256 + t];
        out[(size_t)n * 256 + t] = 1.0f / (1.0f + __expf(-sacc));
    }
}

extern "C" void kernel_launch(void* const* d_in, const int* in_sizes, int n_in,
                              void* d_out, int out_size, void* d_ws, size_t ws_size,
                              hipStream_t stream) {
    (void)in_sizes; (void)n_in; (void)out_size; (void)ws_size;
    const float* src  = (const float*)d_in[0];
    const float* edge = (const float*)d_in[1];
    const float* Wfc  = (const float*)d_in[2];
    const float* bfc  = (const float*)d_in[3];
    const float* Q    = (const float*)d_in[4];
    const float* Wg   = (const float*)d_in[5];
    const float* bg   = (const float*)d_in[6];
    const int*   sidx = (const int*)d_in[7];
    const int*   didx = (const int*)d_in[8];
    float* out = (float*)d_out;

    // workspace layout (floats): scores[E*4] | smax_keys[N*4] | denom[N*4] | deg[N]
    //                            | sum_p[N*256] | pooled[N*4*256]   (~270 MB total)
    float*    ws_scores = (float*)d_ws;
    unsigned* ws_smax   = (unsigned*)(ws_scores + (size_t)N_EDGES * 4);
    float*    ws_denom  = (float*)(ws_smax + (size_t)N_NODES * 4);
    float*    ws_deg    = ws_denom + (size_t)N_NODES * 4;
    float*    ws_sump   = ws_deg + N_NODES;
    float*    ws_pooled = ws_sump + (size_t)N_NODES * 256;

    // zero all accumulators (smax key 0 == encoded -inf identity)
    size_t zero_n = (size_t)N_NODES * (4 + 4 + 1 + 256 + 1024);
    zero_kernel<<<2048, 256, 0, stream>>>((float*)ws_smax, zero_n);

    size_t ldsbytes = (size_t)N_TYPES * LDS_STRIDE * sizeof(_Float16);  // 132 KB of 320 KB WGP LDS
    edge_gemm_scores<<<1024, 128, ldsbytes, stream>>>(
        src, edge, Wfc, bfc, Q, sidx, didx, ws_scores, ws_smax);
    edge_gemm_scatter<<<1024, 128, ldsbytes, stream>>>(
        src, edge, Wfc, bfc, sidx, didx, ws_scores, ws_smax,
        ws_denom, ws_deg, ws_sump, ws_pooled);
    node_kernel<<<(N_NODES + 7) / 8, 256, 0, stream>>>(
        Wg, bg, ws_denom, ws_deg, ws_sump, ws_pooled, out);
}
